// GraphAttention_10642928959904
// MI455X (gfx1250) — matrix operand order repaired and checked by reading, not compiled, not run on hardware
//
#include <hip/hip_runtime.h>
#include <math.h>

// ---------------------------------------------------------------------------
// Problem constants (from reference)
// ---------------------------------------------------------------------------
#define BN      32
#define NPG     200
#define NNODE   6400          // 32*200
#define EALL    32768         // 32 * 2 * 512
#define EK      16384         // every other edge
#define H1C     256
#define H2C     512
#define DATT    1536
#define SEQ     512           // E_DIR rows per batch in attention

// ---------------------------------------------------------------------------
// bf16 helpers / vector types (CDNA5 WMMA operand shapes)
// ---------------------------------------------------------------------------
typedef __attribute__((ext_vector_type(16))) __bf16 v16bf;
typedef __attribute__((ext_vector_type(8)))  __bf16 v8bf;
typedef __attribute__((ext_vector_type(8)))  float  v8f;
typedef __attribute__((ext_vector_type(4)))  float  v4f;

__device__ __forceinline__ __bf16 f2bf(float f) {
  unsigned u = __builtin_bit_cast(unsigned, f);
  u += 0x7FFFu + ((u >> 16) & 1u);            // round-to-nearest-even
  unsigned short h = (unsigned short)(u >> 16);
  return __builtin_bit_cast(__bf16, h);
}
__device__ __forceinline__ float bf2f(__bf16 b) {
  unsigned short h = __builtin_bit_cast(unsigned short, b);
  unsigned u = ((unsigned)h) << 16;
  return __builtin_bit_cast(float, u);
}
// convert 8 fp32 (two float4) -> v8bf
__device__ __forceinline__ v8bf cvt8(const v4f f0, const v4f f1) {
  v8bf r;
#pragma unroll
  for (int i = 0; i < 4; ++i) { r[i] = f2bf(f0[i]); r[4 + i] = f2bf(f1[i]); }
  return r;
}

// Build the 16x32 bf16 A operand row segment: lane L<16 holds row M=L with
// K in {0..15}; lane L+16 holds same row with K in {16..31}.
__device__ __forceinline__ v16bf make_a_operand(const __bf16* row, int half) {
  v8bf a0 = *(const v8bf*)(row + half * 16);
  v8bf a1 = *(const v8bf*)(row + half * 16 + 8);
  v16bf a;
#pragma unroll
  for (int i = 0; i < 8; ++i) { a[i] = a0[i]; a[8 + i] = a1[i]; }
  return a;
}

// ---------------------------------------------------------------------------
// WMMA GEMM:  C[M x N] = op( alpha * (A @ B^T + bias + addend) )
//   A: M x K   (fp32 or bf16, row-major, lda)
//   B: N x K   (fp32 row-major, ldb)  -> B^T column n is B row n (K-contig)
// 64x64 tile per block (128 threads = 4 waves, wave w owns 16 M-rows).
// Fully templated so staging / epilogue are branch-free.
// ---------------------------------------------------------------------------
enum { STORE_F32 = 0, STORE_BF16 = 1, STORE_VT_BF16 = 2 };

template <int A_BF16, int SMODE, int HAS_BIAS, int HAS_ADD, int RELU>
__global__ void __launch_bounds__(128)
wmma_gemm_kernel(const void* __restrict__ Av, int lda,
                 const float* __restrict__ B, int ldb,
                 const float* __restrict__ bias,
                 const float* __restrict__ addend, int ldadd,
                 void* __restrict__ Cv, int ldc,
                 int K, float alpha)
{
  __shared__ __align__(64) __bf16 sA[64][32];
  __shared__ __align__(64) __bf16 sB[64][32];

  const int m0   = blockIdx.x * 64;
  const int n0   = blockIdx.y * 64;
  const int tid  = threadIdx.x;
  const int wave = tid >> 5;
  const int lane = tid & 31;
  const int l16  = lane & 15;
  const int half = lane >> 4;

  // staging assignment: thread owns a contiguous 16-elem half-row
  const int srow = tid >> 1;           // 0..63
  const int scol = (tid & 1) * 16;     // 0 or 16

  v8f acc[4];
#pragma unroll
  for (int j = 0; j < 4; ++j)
#pragma unroll
    for (int r = 0; r < 8; ++r) acc[j][r] = 0.0f;

  for (int k0 = 0; k0 < K; k0 += 32) {
    // ---- stage A tile (64x32) ----
    if (A_BF16) {
      const __bf16* Ab = (const __bf16*)Av + (size_t)(m0 + srow) * lda + k0 + scol;
      v8bf x0 = *(const v8bf*)Ab;
      v8bf x1 = *(const v8bf*)(Ab + 8);
      *(v8bf*)&sA[srow][scol]     = x0;
      *(v8bf*)&sA[srow][scol + 8] = x1;
    } else {
      const float* Af = (const float*)Av + (size_t)(m0 + srow) * lda + k0 + scol;
      v4f f0 = *(const v4f*)Af;
      v4f f1 = *(const v4f*)(Af + 4);
      v4f f2 = *(const v4f*)(Af + 8);
      v4f f3 = *(const v4f*)(Af + 12);
      *(v8bf*)&sA[srow][scol]     = cvt8(f0, f1);
      *(v8bf*)&sA[srow][scol + 8] = cvt8(f2, f3);
    }
    // ---- stage B tile (64x32, always fp32 weights) ----
    {
      const float* Bf = B + (size_t)(n0 + srow) * ldb + k0 + scol;
      v4f f0 = *(const v4f*)Bf;
      v4f f1 = *(const v4f*)(Bf + 4);
      v4f f2 = *(const v4f*)(Bf + 8);
      v4f f3 = *(const v4f*)(Bf + 12);
      *(v8bf*)&sB[srow][scol]     = cvt8(f0, f1);
      *(v8bf*)&sB[srow][scol + 8] = cvt8(f2, f3);
    }
    __syncthreads();

    v16bf av = make_a_operand(&sA[wave * 16 + l16][0], half);
#pragma unroll
    for (int j = 0; j < 4; ++j) {
      // B operand: lane = output column (l16), 16 contiguous K values per half
      v16bf bv = *(const v16bf*)&sB[j * 16 + l16][half * 16];
      acc[j] = __builtin_amdgcn_wmma_f32_16x16x32_bf16(
          false, av, false, bv, (short)0, acc[j], false, false);
    }
    __syncthreads();
  }

  // C/D layout: lane L -> col = l16 (+tile), rows = half*8 + r
  const int rbase = half * 8;
#pragma unroll
  for (int j = 0; j < 4; ++j) {
    int n = n0 + j * 16 + l16;
    float bval = HAS_BIAS ? bias[n] : 0.0f;
#pragma unroll
    for (int r = 0; r < 8; ++r) {
      int m = m0 + wave * 16 + rbase + r;
      float v = acc[j][r] + bval;
      if (HAS_ADD) v += addend[(size_t)m * ldadd + n];
      if (RELU) v = fmaxf(v, 0.0f);
      v *= alpha;
      if (SMODE == STORE_F32) {
        ((float*)Cv)[(size_t)m * ldc + n] = v;
      } else if (SMODE == STORE_BF16) {
        ((__bf16*)Cv)[(size_t)m * ldc + n] = f2bf(v);
      } else {  // STORE_VT_BF16: m = b*512+s, write vT[b][n][s]
        int bb = m >> 9, s = m & 511;
        ((__bf16*)Cv)[((size_t)bb * DATT + n) * SEQ + s] = f2bf(v);
      }
    }
  }
}

// ---------------------------------------------------------------------------
// Per-batch attention: one block = (batch, 16 query rows).
// scores = q@k^T (scale folded into q), softmax fp32, ctx = attn @ v
// ---------------------------------------------------------------------------
__global__ void __launch_bounds__(128)
attention_kernel(const __bf16* __restrict__ q, const __bf16* __restrict__ k,
                 const __bf16* __restrict__ vT, __bf16* __restrict__ ctx)
{
  __shared__ float              sScores[16][SEQ];   // 32 KB
  __shared__ __align__(64) __bf16 sAttn[16][SEQ];   // 16 KB

  const int b    = blockIdx.y;
  const int m0   = blockIdx.x * 16;
  const int tid  = threadIdx.x;
  const int wave = tid >> 5;
  const int lane = tid & 31;
  const int l16  = lane & 15;
  const int half = lane >> 4;

  const __bf16* qb = q  + (size_t)b * SEQ * DATT;
  const __bf16* kb = k  + (size_t)b * SEQ * DATT;
  const __bf16* vb = vT + (size_t)b * DATT * SEQ;

  // ---- phase 1: scores (each wave owns 128 keys = 8 tiles) ----
  v8f acc[8];
#pragma unroll
  for (int j = 0; j < 8; ++j)
#pragma unroll
    for (int r = 0; r < 8; ++r) acc[j][r] = 0.0f;

  for (int ks = 0; ks < DATT / 32; ++ks) {
    v16bf av = make_a_operand(qb + (size_t)(m0 + l16) * DATT + ks * 32, half);
#pragma unroll
    for (int j = 0; j < 8; ++j) {
      int key = wave * 128 + j * 16 + l16;
      v16bf bv = *(const v16bf*)(kb + (size_t)key * DATT + ks * 32 + half * 16);
      acc[j] = __builtin_amdgcn_wmma_f32_16x16x32_bf16(
          false, av, false, bv, (short)0, acc[j], false, false);
    }
  }
#pragma unroll
  for (int j = 0; j < 8; ++j)
#pragma unroll
    for (int r = 0; r < 8; ++r)
      sScores[half * 8 + r][wave * 128 + j * 16 + l16] = acc[j][r];
  __syncthreads();

  // ---- phase 2: fp32 softmax over 512 keys (scale already in q) ----
  if (tid < 16) {
    float mx = -3.0e38f;
    for (int s = 0; s < SEQ; ++s) mx = fmaxf(mx, sScores[tid][s]);
    float sum = 0.0f;
    for (int s = 0; s < SEQ; ++s) {
      float e = __expf(sScores[tid][s] - mx);
      sScores[tid][s] = e;
      sum += e;
    }
    float inv = 1.0f / sum;
    for (int s = 0; s < SEQ; ++s) sAttn[tid][s] = f2bf(sScores[tid][s] * inv);
  }
  __syncthreads();

  // ---- phase 3: ctx = attn @ v (each wave owns 24 of 96 d-tiles) ----
  for (int i = 0; i < 24; ++i) {
    int dt = wave * 24 + i;
    v8f c;
#pragma unroll
    for (int r = 0; r < 8; ++r) c[r] = 0.0f;
    for (int st = 0; st < SEQ / 32; ++st) {
      v16bf av = make_a_operand(&sAttn[l16][st * 32], half);
      v16bf bv = *(const v16bf*)(vb + (size_t)(dt * 16 + l16) * SEQ +
                                 st * 32 + half * 16);
      c = __builtin_amdgcn_wmma_f32_16x16x32_bf16(
          false, av, false, bv, (short)0, c, false, false);
    }
#pragma unroll
    for (int r = 0; r < 8; ++r)
      ctx[((size_t)b * SEQ + m0 + half * 8 + r) * DATT + dt * 16 + l16] =
          f2bf(c[r]);
  }
}

// ---------------------------------------------------------------------------
// Glue kernels
// ---------------------------------------------------------------------------
__global__ void edge_mlp_kernel(const float* __restrict__ x,
                                const int* __restrict__ esrc,
                                const int* __restrict__ edst,
                                const float* __restrict__ eattr,
                                const float* __restrict__ W1,
                                const float* __restrict__ b1,
                                const float* __restrict__ W2,
                                const float* __restrict__ b2,
                                float* __restrict__ w_edge)
{
  int e = blockIdx.x * blockDim.x + threadIdx.x;
  if (e >= EALL) return;
  int s = esrc[e], d = edst[e];
  float w = eattr[2 * e], cl = eattr[2 * e + 1];
  float emb[7] = { x[s * 5], x[s * 5 + 1], w, 1.0f - cl, cl,
                   x[d * 5], x[d * 5 + 1] };
  float acc2 = 0.0f;
  for (int o = 0; o < 64; ++o) {
    float h = b1[o];
#pragma unroll
    for (int kk = 0; kk < 7; ++kk) h += W1[o * 7 + kk] * emb[kk];
    acc2 += W2[o] * fmaxf(h, 0.0f);
  }
  w_edge[e] = fmaxf(acc2 + b2[0], 0.0f);
}

__global__ void scatter1_kernel(const float* __restrict__ x,
                                const int* __restrict__ esrc,
                                const int* __restrict__ edst,
                                const float* __restrict__ w_edge,
                                float* __restrict__ agg)
{
  int e = blockIdx.x * blockDim.x + threadIdx.x;
  if (e >= EALL) return;
  int s = esrc[e], d = edst[e];
  float w = w_edge[e];
#pragma unroll
  for (int j = 0; j < 5; ++j) atomicAdd(&agg[d * 5 + j], w * x[s * 5 + j]);
}

__global__ void conv1_kernel(const float* __restrict__ agg,
                             const float* __restrict__ x,
                             const float* __restrict__ Wrel,
                             const float* __restrict__ brel,
                             const float* __restrict__ Wroot,
                             float* __restrict__ h1)
{
  int idx = blockIdx.x * 256 + threadIdx.x;     // 6400*256 threads
  int n = idx >> 8, o = idx & 255;
  float acc = brel[o];
#pragma unroll
  for (int kk = 0; kk < 5; ++kk)
    acc += agg[n * 5 + kk] * Wrel[o * 5 + kk] + x[n * 5 + kk] * Wroot[o * 5 + kk];
  h1[(size_t)n * H1C + o] = fmaxf(acc, 0.0f);
}

__global__ void scatter2_kernel(const float* __restrict__ h1,
                                const int* __restrict__ esrc,
                                const int* __restrict__ edst,
                                const float* __restrict__ w_edge,
                                float* __restrict__ agg2)
{
  int e = blockIdx.x;                            // one block per edge
  int c = threadIdx.x;                           // 256 channels
  float w = w_edge[e];
  atomicAdd(&agg2[(size_t)edst[e] * H1C + c], w * h1[(size_t)esrc[e] * H1C + c]);
}

__global__ void gn_stats_kernel(const float* __restrict__ x2,
                                const float* __restrict__ gms,
                                float* __restrict__ mean,
                                float* __restrict__ invstd)
{
  int b = blockIdx.x, c = threadIdx.x;           // 32 blocks x 512
  float s = 0.0f;
  for (int i = 0; i < NPG; ++i) s += x2[((size_t)(b * NPG + i)) * H2C + c];
  float m = s * (1.0f / NPG);
  float ms = m * gms[c];
  float v = 0.0f;
  for (int i = 0; i < NPG; ++i) {
    float ce = x2[((size_t)(b * NPG + i)) * H2C + c] - ms;
    v += ce * ce;
  }
  mean[b * H2C + c] = m;
  invstd[b * H2C + c] = rsqrtf(v * (1.0f / NPG) + 1e-5f);
}

__global__ void gn_apply_kernel(const float* __restrict__ x2,
                                const float* __restrict__ mean,
                                const float* __restrict__ invstd,
                                const float* __restrict__ gms,
                                const float* __restrict__ gw,
                                const float* __restrict__ gb,
                                float* __restrict__ xn)
{
  size_t idx = (size_t)blockIdx.x * 256 + threadIdx.x;   // 6400*512
  int c = (int)(idx & (H2C - 1));
  int n = (int)(idx >> 9);
  int b = n / NPG;
  float ce = x2[idx] - mean[b * H2C + c] * gms[c];
  xn[idx] = gw[c] * ce * invstd[b * H2C + c] + gb[c];
}

__global__ void sort_kernel(const int* __restrict__ esrc,
                            const int* __restrict__ edst,
                            const float* __restrict__ eattr,
                            int* __restrict__ se_src, int* __restrict__ se_dst,
                            float* __restrict__ se_w, float* __restrict__ se_cls)
{
  __shared__ int sk[SEQ];
  __shared__ int sv[SEQ];
  int b = blockIdx.x, t = threadIdx.x;           // 32 blocks x 512
  int eidx = b * 1024 + 2 * t + 1;               // edges[:,1::2] within batch
  sk[t] = esrc[eidx] * NNODE + edst[eidx];
  sv[t] = eidx;
  __syncthreads();
  for (int ksz = 2; ksz <= SEQ; ksz <<= 1)
    for (int j = ksz >> 1; j > 0; j >>= 1) {
      int ixj = t ^ j;
      if (ixj > t) {
        bool up = (t & ksz) == 0;
        if ((sk[t] > sk[ixj]) == up) {
          int tk = sk[t]; sk[t] = sk[ixj]; sk[ixj] = tk;
          int tv = sv[t]; sv[t] = sv[ixj]; sv[ixj] = tv;
        }
      }
      __syncthreads();
    }
  int e2 = sv[t];
  int g = b * SEQ + t;
  se_src[g] = esrc[e2];
  se_dst[g] = edst[e2];
  se_w[g]   = eattr[2 * e2];
  se_cls[g] = eattr[2 * e2 + 1];
}

__global__ void feat_kernel(const float* __restrict__ xn,
                            const int* __restrict__ se_src,
                            const int* __restrict__ se_dst,
                            const float* __restrict__ se_w,
                            const float* __restrict__ se_cls,
                            const float* __restrict__ Wee,
                            const float* __restrict__ bee,
                            __bf16* __restrict__ feat)
{
  size_t idx = (size_t)blockIdx.x * 256 + threadIdx.x;    // EK*DATT
  int d = (int)(idx % DATT);
  int g = (int)(idx / DATT);
  float v;
  if (d < 512) {
    v = xn[(size_t)se_src[g] * H2C + d];
  } else if (d < 1024) {
    int j = d - 512;
    float w = se_w[g], cl = se_cls[g];
    v = fmaxf(Wee[j * 3] * w + Wee[j * 3 + 1] * (1.0f - cl) +
              Wee[j * 3 + 2] * cl + bee[j], 0.0f);
  } else {
    v = xn[(size_t)se_dst[g] * H2C + (d - 1024)];
  }
  feat[idx] = f2bf(v);
}

__global__ void transpose_kernel(const float* __restrict__ in,
                                 float* __restrict__ out)  // 4608x1536 -> 1536x4608
{
  size_t idx = (size_t)blockIdx.x * 256 + threadIdx.x;
  int i = (int)(idx / DATT), j = (int)(idx % DATT);
  out[(size_t)j * 4608 + i] = in[idx];
}

__global__ void beff_kernel(const float* __restrict__ W_in,
                            const float* __restrict__ b_in,
                            const float* __restrict__ b_qkv,
                            float* __restrict__ beff)
{
  int idx = blockIdx.x * 256 + threadIdx.x;
  if (idx >= 3 * DATT) return;
  int sec = idx / DATT;
  float s = b_in[idx];
  const float* wrow = W_in + (size_t)idx * DATT;
  const float* bq = b_qkv + sec * DATT;
  for (int m = 0; m < DATT; ++m) s += wrow[m] * bq[m];
  beff[idx] = s;
}

__global__ void wcomb_kernel(const float* __restrict__ W_out,
                             const float* __restrict__ W_o,
                             const float* __restrict__ b_o,
                             const float* __restrict__ b_out,
                             float* __restrict__ wcomb)
{
  int j = blockIdx.x * 256 + threadIdx.x;
  if (j >= DATT) return;
  float s = 0.0f;
  for (int i = 0; i < DATT; ++i) s += W_out[i] * W_o[(size_t)i * DATT + j];
  wcomb[j] = s;
  if (j == 0) {
    float bb = b_out[0];
    for (int i = 0; i < DATT; ++i) bb += W_out[i] * b_o[i];
    wcomb[DATT] = bb;
  }
}

__global__ void out_kernel(const __bf16* __restrict__ ctx,
                           const float* __restrict__ wcomb,
                           float* __restrict__ outsc)
{
  int g = blockIdx.x * 8 + (threadIdx.x >> 5);   // one wave32 per row
  int lane = threadIdx.x & 31;
  const __bf16* row = ctx + (size_t)g * DATT;
  float s = 0.0f;
  for (int i = lane; i < DATT; i += 32) s += bf2f(row[i]) * wcomb[i];
#pragma unroll
  for (int off = 16; off; off >>= 1) s += __shfl_down(s, off, 32);
  if (lane == 0) outsc[g] = s + wcomb[DATT];
}

__global__ void final_kernel(const float* __restrict__ outsc,
                             const int* __restrict__ se_src,
                             const int* __restrict__ se_dst,
                             const float* __restrict__ se_cls,
                             float* __restrict__ dout)
{
  int idx = blockIdx.x * 256 + threadIdx.x;      // 32*256 pairs
  if (idx >= BN * 256) return;
  int b = idx >> 8, i = idx & 255;
  int g0 = b * SEQ + 2 * i;
  float v0 = outsc[g0], v1 = outsc[g0 + 1];
  int sel = (v0 <= v1) ? 0 : 1;
  dout[idx * 2]     = (float)se_src[g0];          // e_stack[:,0::2,:]
  dout[idx * 2 + 1] = (float)se_dst[g0];
  dout[BN * 256 * 2 + idx] = fminf(v0, v1);       // min_feat
  dout[BN * 256 * 3 + idx] = se_cls[g0 + sel];    // min_cls
}

// ---------------------------------------------------------------------------
// Host orchestration
// ---------------------------------------------------------------------------
static inline size_t alignup(size_t x) { return (x + 255) & ~(size_t)255; }

extern "C" void kernel_launch(void* const* d_in, const int* in_sizes, int n_in,
                              void* d_out, int out_size, void* d_ws, size_t ws_size,
                              hipStream_t stream) {
  (void)in_sizes; (void)n_in; (void)out_size; (void)ws_size;

  const float* x       = (const float*)d_in[0];
  const int*   edges   = (const int*)d_in[1];
  const float* eattr   = (const float*)d_in[2];
  const float* W_emb1  = (const float*)d_in[5];
  const float* b_emb1  = (const float*)d_in[6];
  const float* W_emb2  = (const float*)d_in[7];
  const float* b_emb2  = (const float*)d_in[8];
  const float* Wrel1   = (const float*)d_in[9];
  const float* brel1   = (const float*)d_in[10];
  const float* Wroot1  = (const float*)d_in[11];
  const float* Wrel2   = (const float*)d_in[12];
  const float* brel2   = (const float*)d_in[13];
  const float* Wroot2  = (const float*)d_in[14];
  const float* gn_w    = (const float*)d_in[15];
  const float* gn_b    = (const float*)d_in[16];
  const float* gn_ms   = (const float*)d_in[17];
  const float* W_ee    = (const float*)d_in[18];
  const float* b_ee    = (const float*)d_in[19];
  const float* W_qkv   = (const float*)d_in[20];
  const float* b_qkv   = (const float*)d_in[21];
  const float* W_in    = (const float*)d_in[22];
  const float* b_in    = (const float*)d_in[23];
  const float* W_o     = (const float*)d_in[24];
  const float* b_o     = (const float*)d_in[25];
  const float* W_out   = (const float*)d_in[26];
  const float* b_out   = (const float*)d_in[27];

  const int* esrc = edges;
  const int* edst = edges + EALL;

  // workspace carve-up
  char* p = (char*)d_ws;
  size_t off = 0;
  auto carve = [&](size_t bytes) { void* r = p + off; off = alignup(off + bytes); return r; };

  float* w_edge = (float*)carve(EALL * 4);
  float* agg1   = (float*)carve((size_t)NNODE * 5 * 4);
  float* h1     = (float*)carve((size_t)NNODE * H1C * 4);
  float* agg2   = (float*)carve((size_t)NNODE * H1C * 4);
  float* tmp2   = (float*)carve((size_t)NNODE * H2C * 4);
  float* x2     = (float*)carve((size_t)NNODE * H2C * 4);
  float* xn     = (float*)carve((size_t)NNODE * H2C * 4);
  float* meanb  = (float*)carve((size_t)BN * H2C * 4);
  float* invstd = (float*)carve((size_t)BN * H2C * 4);
  int*   se_src = (int*)carve((size_t)EK * 4);
  int*   se_dst = (int*)carve((size_t)EK * 4);
  float* se_w   = (float*)carve((size_t)EK * 4);
  float* se_cls = (float*)carve((size_t)EK * 4);
  float* WqkvT  = (float*)carve((size_t)DATT * 3 * DATT * 4);   // 1536 x 4608
  float* Weff   = (float*)carve((size_t)3 * DATT * DATT * 4);
  float* beff   = (float*)carve((size_t)3 * DATT * 4);
  float* wcomb  = (float*)carve((size_t)(DATT + 1) * 4);
  __bf16* feat  = (__bf16*)carve((size_t)EK * DATT * 2);
  __bf16* qbuf  = (__bf16*)carve((size_t)EK * DATT * 2);
  __bf16* kbuf  = (__bf16*)carve((size_t)EK * DATT * 2);
  __bf16* vT    = (__bf16*)carve((size_t)EK * DATT * 2);
  __bf16* ctx   = (__bf16*)carve((size_t)EK * DATT * 2);
  float* outsc  = (float*)carve((size_t)EK * 4);

  hipMemsetAsync(agg1, 0, (size_t)NNODE * 5 * 4, stream);
  hipMemsetAsync(agg2, 0, (size_t)NNODE * H1C * 4, stream);

  // 1) edge weight MLP
  edge_mlp_kernel<<<EALL / 256, 256, 0, stream>>>(x, esrc, edst, eattr,
      W_emb1, b_emb1, W_emb2, b_emb2, w_edge);

  // 2) conv1
  scatter1_kernel<<<EALL / 256, 256, 0, stream>>>(x, esrc, edst, w_edge, agg1);
  conv1_kernel<<<(NNODE * H1C) / 256, 256, 0, stream>>>(agg1, x, Wrel1, brel1,
      Wroot1, h1);

  // 3) conv2: scatter + two WMMA GEMMs (6400x256 @ 256x512)
  scatter2_kernel<<<EALL, 256, 0, stream>>>(h1, esrc, edst, w_edge, agg2);
  wmma_gemm_kernel<0, STORE_F32, 0, 0, 0>
      <<<dim3(NNODE / 64, H2C / 64), 128, 0, stream>>>(
      h1, H1C, Wroot2, H1C, nullptr, nullptr, 0, tmp2, H2C, H1C, 1.0f);
  wmma_gemm_kernel<0, STORE_F32, 1, 1, 1>
      <<<dim3(NNODE / 64, H2C / 64), 128, 0, stream>>>(
      agg2, H1C, Wrel2, H1C, brel2, tmp2, H2C, x2, H2C, H1C, 1.0f);

  // 4) group norm
  gn_stats_kernel<<<BN, H2C, 0, stream>>>(x2, gn_ms, meanb, invstd);
  gn_apply_kernel<<<(NNODE * H2C) / 256, 256, 0, stream>>>(x2, meanb, invstd,
      gn_ms, gn_w, gn_b, xn);

  // 5) per-batch bitonic sort of the 512 k-edges
  sort_kernel<<<BN, SEQ, 0, stream>>>(esrc, edst, eattr, se_src, se_dst, se_w,
      se_cls);

  // 6) combined projection weights: Weff_s = W_in_s @ W_qkv_s  (fold qkv GEMM)
  transpose_kernel<<<(3 * DATT * DATT) / 256, 256, 0, stream>>>(W_qkv, WqkvT);
  for (int s = 0; s < 3; ++s) {
    wmma_gemm_kernel<0, STORE_F32, 0, 0, 0>
        <<<dim3(DATT / 64, DATT / 64), 128, 0, stream>>>(
        W_in + (size_t)s * DATT * DATT, DATT,
        WqkvT + (size_t)s * DATT, 3 * DATT, nullptr, nullptr, 0,
        Weff + (size_t)s * DATT * DATT, DATT, DATT, 1.0f);
  }
  beff_kernel<<<(3 * DATT + 255) / 256, 256, 0, stream>>>(W_in, b_in, b_qkv, beff);
  wcomb_kernel<<<(DATT + 255) / 256, 256, 0, stream>>>(W_out, W_o, b_o, b_out,
      wcomb);

  // 7) feature assembly (bf16)
  feat_kernel<<<((size_t)EK * DATT) / 256, 256, 0, stream>>>(xn, se_src, se_dst,
      se_w, se_cls, W_ee, b_ee, feat);

  // 8) q/k/v projections (16384x1536 @ 1536x1536, bf16 A, WMMA)
  const float inv_sqrt_d = 1.0f / sqrtf((float)DATT);
  wmma_gemm_kernel<1, STORE_BF16, 1, 0, 0>
      <<<dim3(EK / 64, DATT / 64), 128, 0, stream>>>(
      feat, DATT, Weff + 0 * (size_t)DATT * DATT, DATT, beff + 0 * DATT,
      nullptr, 0, qbuf, DATT, DATT, inv_sqrt_d);
  wmma_gemm_kernel<1, STORE_BF16, 1, 0, 0>
      <<<dim3(EK / 64, DATT / 64), 128, 0, stream>>>(
      feat, DATT, Weff + 1 * (size_t)DATT * DATT, DATT, beff + 1 * DATT,
      nullptr, 0, kbuf, DATT, DATT, 1.0f);
  wmma_gemm_kernel<1, STORE_VT_BF16, 1, 0, 0>
      <<<dim3(EK / 64, DATT / 64), 128, 0, stream>>>(
      feat, DATT, Weff + 2 * (size_t)DATT * DATT, DATT, beff + 2 * DATT,
      nullptr, 0, vT, DATT, DATT, 1.0f);

  // 9) attention (WMMA scores + softmax + WMMA context)
  attention_kernel<<<dim3(SEQ / 16, BN), 128, 0, stream>>>(qbuf, kbuf, vT, ctx);

  // 10) folded output head: out = ctx . (W_out @ W_o) + const
  out_kernel<<<EK / 8, 256, 0, stream>>>(ctx, wcomb, outsc);

  // 11) pairwise min / argmin + edge index outputs
  final_kernel<<<(BN * 256) / 256, 256, 0, stream>>>(outsc, se_src, se_dst,
      se_cls, (float*)d_out);
}
